// AttentionPoolingV3_79551384256833
// MI455X (gfx1250) — matrix operand order repaired and checked by reading, not compiled, not run on hardware
//
#include <hip/hip_runtime.h>
#include <math.h>

// ---------------------------------------------------------------------------
// AttentionPoolingV3 for gfx1250 (MI455X, wave32, WMMA + TDM)
//   scores = tanh(x @ W1 + b1) @ W2 + b2        (bf16 WMMA GEMM, f32 accum)
//   per-segment softmax over sorted batch ids, weighted mean of x
// ---------------------------------------------------------------------------

#define D_DIM 256
#define H_DIM 128
#define NSEG  1024
#define LDS_ROW_B 272                    // 128 bf16 (256B) + 16B pad per row
#define LDS_W1_BYTES (256 * LDS_ROW_B)   // 69632 bytes

typedef __attribute__((ext_vector_type(16))) __bf16        v16bf;
typedef __attribute__((ext_vector_type(8)))  float         v8f;
typedef __attribute__((ext_vector_type(4)))  unsigned int  u32x4;
typedef __attribute__((ext_vector_type(8)))  int           i32x8;
typedef __attribute__((ext_vector_type(4)))  int           i32x4;

union BfVec {
    unsigned int u[8];
    v16bf        v;
};

#if defined(__has_builtin)
#  if __has_builtin(__builtin_amdgcn_tensor_load_to_lds)
#    define HAVE_TDM 1
#  endif
#endif
#ifndef HAVE_TDM
#  define HAVE_TDM 0
#endif

#define LOG2E      1.4426950408889634f
#define TWO_LOG2E  2.8853900817779268f

// hardware-transcendental fast math: one v_exp_f32 / v_rcp_f32 each
__device__ __forceinline__ float exp_fast(float v) {
    return __builtin_amdgcn_exp2f(v * LOG2E);
}
// tanh(x) = 1 - 2/(exp(2x)+1); saturates correctly for |x| large
__device__ __forceinline__ float tanh_fast(float v) {
    float e = __builtin_amdgcn_exp2f(v * TWO_LOG2E);
    return 1.0f - 2.0f * __builtin_amdgcn_rcpf(e + 1.0f);
}

// RNE conversion for the one-time W1 conversion kernel
__device__ __forceinline__ unsigned short f32_to_bf16_rne(float f) {
    unsigned int u = __float_as_uint(f);
    unsigned int r = 0x7fffu + ((u >> 16) & 1u);
    return (unsigned short)((u + r) >> 16);
}

// Fast pack for the hot GEMM path: round-to-nearest (ties up) by adding
// 0x8000 to the raw bits, then one v_perm_b32 packs both high halves.
// dst = { bf16(hi) , bf16(lo) }
__device__ __forceinline__ unsigned int pack2_bf16(float lo, float hi) {
    unsigned int ulo = __float_as_uint(lo) + 0x8000u;
    unsigned int uhi = __float_as_uint(hi) + 0x8000u;
    return __builtin_amdgcn_perm(uhi, ulo, 0x07060302u);
}

// ---------------------------------------------------------------------------
// Kernel 0: convert W1 (256x128 f32, row-major) to bf16 row-major in ws.
// ---------------------------------------------------------------------------
__global__ void __launch_bounds__(256) convert_w1(const float* __restrict__ W1,
                                                  unsigned short* __restrict__ W1b) {
    int i = blockIdx.x * 256 + threadIdx.x;
    if (i < D_DIM * H_DIM) W1b[i] = f32_to_bf16_rne(W1[i]);
}

// ---------------------------------------------------------------------------
// Kernel 1: scores[i] = tanh(x_i @ W1 + b1) @ W2 + b2  via bf16 WMMA.
// One wave = 16 rows: 8 K-tiles x 8 N-tiles = 64 v_wmma per wave.
//
// W1(bf16) staged into LDS once per block by the Tensor Data Mover with a
// 16B/row pad (row pitch 272B = 68 dwords -> 2-way bank conflicts max on
// the per-lane B reads, vs 32-way at a 256B pitch).
//
// A fragment (16x32 bf16, M x K): lane L -> M = L%16, kg = L/16.
//   elements 0..7  : K = kg*8 + 0..7      elements 8..15 : K = 16+kg*8+0..7
// B fragment (32x16 bf16, K x N, column-major convention):
//   lane L -> N = L%16 ; elements 0..7 -> K = kg*8+0..7, 8..15 -> K = 16+kg*8+0..7 (LDS column slice)
// C/D (16x16 f32): VGPR r -> M = r + 8*(L/16); lane -> N = L%16.
// ---------------------------------------------------------------------------
__global__ void __launch_bounds__(256) scores_kernel(
        const float* __restrict__ x,
        const unsigned short* __restrict__ W1b,
        const float* __restrict__ b1,
        const float* __restrict__ W2,
        const float* __restrict__ b2,
        float* __restrict__ s,
        int N) {
    extern __shared__ char smem[];   // LDS_W1_BYTES: padded bf16 W1

    // ---- stage W1 into LDS ----
#if HAVE_TDM
    if (threadIdx.x < 32) {          // wave 0 issues one TDM descriptor
        unsigned int       lds_base = (unsigned int)(unsigned long long)(void*)smem;
        unsigned long long ga       = (unsigned long long)(const void*)W1b;

        u32x4 g0;
        g0[0] = 1u;                                   // count=1, user mode
        g0[1] = lds_base;                             // lds_addr
        g0[2] = (unsigned int)ga;                     // global_addr[31:0]
        g0[3] = (unsigned int)((ga >> 32) & 0x01FFFFFFu) | 0x80000000u; // addr[56:32] | type=2

        i32x8 g1;
        // data_size=1 (2B) | pad_enable | pad_interval=64dw (code 5) | pad_amount=4dw (code 3)
        g1[0] = (int)((1u << 16) | (1u << 20) | (5u << 22) | (3u << 25));
        g1[1] = (int)(128u << 16);   // tensor_dim0 = 128 (bits 79:48, low half)
        g1[2] = (int)(256u << 16);   // tensor_dim1 = 256 (bits 111:80, low half)
        g1[3] = (int)(128u << 16);   // tile_dim0   = 128 (bits 127:112)
        g1[4] = 256;                 // tile_dim1   = 256 (bits 143:128)
        g1[5] = 128;                 // tensor_dim0_stride = 128 (bits 207:160, low)
        g1[6] = 0;
        g1[7] = 0;

        i32x4 gz = (i32x4)0;         // groups 2/3 unused (2D tensor)
#if defined(__clang_major__) && (__clang_major__ >= 23)
        i32x8 gz8 = (i32x8)0;
        __builtin_amdgcn_tensor_load_to_lds(g0, g1, gz, gz, gz8, 0);
#else
        __builtin_amdgcn_tensor_load_to_lds(g0, g1, gz, gz, 0);
#endif
        __builtin_amdgcn_s_wait_tensorcnt(0);
    }
    __syncthreads();
#else
    // fallback: manual padded fill (16B chunks)
    for (int i = threadIdx.x; i < 2048; i += 256) {
        int row = i >> 3, seg = i & 7;
        uint4 v = ((const uint4*)W1b)[i];
        *(uint4*)(smem + row * LDS_ROW_B + seg * 16) = v;
    }
    __syncthreads();
#endif

    const int  lane    = threadIdx.x & 31;
    const int  wave    = threadIdx.x >> 5;
    const int  tile    = blockIdx.x * 8 + wave;
    const long rowbase = (long)tile * 16;
    const bool active  = (rowbase < N);

    const int m16 = lane & 15;
    const int kg  = lane >> 4;

    if (active) {
        long arow = rowbase + m16;
        if (arow >= N) arow = N - 1;         // tail clamp (N%16==0 here)
        const float* xrow = x + arow * D_DIM;

        v8f C[8];
#pragma unroll
        for (int nt = 0; nt < 8; ++nt) C[nt] = (v8f)(0.0f);

#pragma unroll
        for (int kt = 0; kt < 8; ++kt) {
            // ---- A fragment: 2 x 32B contiguous f32 loads, perm-pack ----
            const float4* ap = (const float4*)(xrow + kt * 32 + kg * 8);
            float4 a0 = ap[0];
            float4 a1 = ap[1];
            float4 a2 = ap[4];   // +16 floats
            float4 a3 = ap[5];
            BfVec A;
            A.u[0] = pack2_bf16(a0.x, a0.y);
            A.u[1] = pack2_bf16(a0.z, a0.w);
            A.u[2] = pack2_bf16(a1.x, a1.y);
            A.u[3] = pack2_bf16(a1.z, a1.w);
            A.u[4] = pack2_bf16(a2.x, a2.y);
            A.u[5] = pack2_bf16(a2.z, a2.w);
            A.u[6] = pack2_bf16(a3.x, a3.y);
            A.u[7] = pack2_bf16(a3.z, a3.w);

#pragma unroll
            for (int nt = 0; nt < 8; ++nt) {
                // ---- B fragment from padded LDS: lane -> N = L%16; elems 0..7 K = kg*8+0..7, 8..15 K = 16+kg*8+0..7 ----
                const unsigned short* bp = (const unsigned short*)(smem + (kt * 32 + kg * 8) * LDS_ROW_B + (nt * 16 + m16) * 2);
                BfVec B;
#pragma unroll
                for (int j = 0; j < 8; ++j) {
                    const int r0 = (j < 4) ? (2 * j) : (2 * j + 8);
                    B.u[j] = (unsigned int)bp[r0 * 136] | ((unsigned int)bp[(r0 + 1) * 136] << 16);
                }

                C[nt] = __builtin_amdgcn_wmma_f32_16x16x32_bf16(
                    false, A.v, false, B.v, (short)0, C[nt], false, false);
            }
        }

        // ---- epilogue: tanh(h + b1) * W2, reduce over N, + b2 ----
        float sacc[8];
#pragma unroll
        for (int r = 0; r < 8; ++r) sacc[r] = 0.0f;

#pragma unroll
        for (int nt = 0; nt < 8; ++nt) {
            const int col = nt * 16 + m16;
            const float bb = b1[col];
            const float w2 = W2[col];
#pragma unroll
            for (int r = 0; r < 8; ++r)
                sacc[r] += tanh_fast(C[nt][r] + bb) * w2;
        }

        const float bias2 = b2[0];
#pragma unroll
        for (int r = 0; r < 8; ++r) {
            float v = sacc[r];
            v += __shfl_xor(v, 1, 32);   // butterfly within each 16-lane half
            v += __shfl_xor(v, 2, 32);
            v += __shfl_xor(v, 4, 32);
            v += __shfl_xor(v, 8, 32);
            if (m16 == 0) {
                long rr = rowbase + r + 8 * kg;
                if (rr < N) s[rr] = v + bias2;
            }
        }
    }
}

// ---------------------------------------------------------------------------
// Kernel 2: per-segment softmax + weighted mean. batch is sorted, so each
// block binary-searches its contiguous row range; fully atomic-free.
// ---------------------------------------------------------------------------
__device__ __forceinline__ int lower_bound_i(const int* __restrict__ a,
                                             int n, int key) {
    int lo = 0, hi = n;
    while (lo < hi) {
        int mid = (lo + hi) >> 1;
        if (a[mid] < key) lo = mid + 1; else hi = mid;
    }
    return lo;
}

__global__ void __launch_bounds__(256) pool_kernel(
        const float* __restrict__ x,
        const int* __restrict__ batch,
        const float* __restrict__ s,
        float* __restrict__ out,
        int N) {
    const int seg = blockIdx.x;
    const int t   = threadIdx.x;

    __shared__ float red[256];
    __shared__ float wbuf[256];

    const int lo = lower_bound_i(batch, N, seg);
    const int hi = lower_bound_i(batch, N, seg + 1);

    // ---- segment max ----
    float m = -INFINITY;
    for (int i = lo + t; i < hi; i += 256) m = fmaxf(m, s[i]);
    red[t] = m;
    __syncthreads();
    for (int off = 128; off > 0; off >>= 1) {
        if (t < off) red[t] = fmaxf(red[t], red[t + off]);
        __syncthreads();
    }
    m = red[0];
    __syncthreads();

    // ---- softmax denominator ----
    float d = 0.0f;
    for (int i = lo + t; i < hi; i += 256) d += exp_fast(s[i] - m);
    red[t] = d;
    __syncthreads();
    for (int off = 128; off > 0; off >>= 1) {
        if (t < off) red[t] += red[t + off];
        __syncthreads();
    }
    const float denom = red[0];
    __syncthreads();

    // ---- weighted accumulation: thread t owns column t ----
    float acc = 0.0f;
    for (int base = lo; base < hi; base += 256) {
        int i = base + t;
        wbuf[t] = (i < hi) ? exp_fast(s[i] - m) : 0.0f;
        __syncthreads();
        if (base + 256 < hi)   // hint next chunk into cache
            __builtin_prefetch(x + (long)(base + 256) * D_DIM + t, 0, 1);
        const int cnt = min(256, hi - base);
        for (int j = 0; j < cnt; ++j)
            acc += x[(long)(base + j) * D_DIM + t] * wbuf[j];
        __syncthreads();
    }

    float r = 0.0f;
    if (hi > lo) {
        const float cntf = (float)(hi - lo);
        r = acc / denom / fmaxf(cntf, 1.0f);
    }
    out[seg * D_DIM + t] = r;
}

// ---------------------------------------------------------------------------
// Launch
// ---------------------------------------------------------------------------
extern "C" void kernel_launch(void* const* d_in, const int* in_sizes, int n_in,
                              void* d_out, int out_size, void* d_ws, size_t ws_size,
                              hipStream_t stream) {
    const float* x     = (const float*)d_in[0];
    const int*   batch = (const int*)d_in[1];
    const float* W1    = (const float*)d_in[2];
    const float* b1    = (const float*)d_in[3];
    const float* W2    = (const float*)d_in[4];
    const float* b2    = (const float*)d_in[5];
    float* out = (float*)d_out;
    const int N = in_sizes[1];

    // allow > 64 KB dynamic LDS for the padded W1 stage (host-side, capture-safe)
    (void)hipFuncSetAttribute((const void*)scores_kernel,
                              hipFuncAttributeMaxDynamicSharedMemorySize,
                              LDS_W1_BYTES);

    // workspace layout: [0, 64KB) bf16 W1 ; [64KB, 64KB + 4N) scores
    unsigned short* W1b = (unsigned short*)d_ws;
    float* s = (float*)((char*)d_ws + 64 * 1024);

    convert_w1<<<dim3((D_DIM * H_DIM + 255) / 256), dim3(256), 0, stream>>>(W1, W1b);

    const int ntiles  = (N + 15) / 16;
    const int nblocks = (ntiles + 7) / 8;   // 8 waves (row tiles) per block
    scores_kernel<<<dim3(nblocks), dim3(256), 0, stream>>>(x, W1b, b1, W2, b2, s, N);

    pool_kernel<<<dim3(NSEG), dim3(256), 0, stream>>>(x, batch, s, out, N);
}
